// SelfAttention_86346022519290
// MI455X (gfx1250) — matrix operand order repaired and checked by reading, not compiled, not run on hardware
//
#include <hip/hip_runtime.h>
#include <stdint.h>

// ---- shapes -----------------------------------------------------------------
constexpr int Bsz  = 128;
constexpr int Nseq = 256;
constexpr int Dim  = 256;
constexpr int Hdim = 256;

// ---- types ------------------------------------------------------------------
typedef __attribute__((ext_vector_type(16))) __bf16  bf16x16;
typedef __attribute__((ext_vector_type(8)))  float   floatx8;
typedef __attribute__((ext_vector_type(4)))  unsigned int u32x4;

union Frag {              // 16 bf16 = one WMMA A/B operand (8 VGPRs)
  bf16x16  v;
  u32x4    q[2];
  uint16_t u[16];
};

__device__ __forceinline__ uint16_t f2bf(float f) {   // RNE fp32 -> bf16
  uint32_t u = __float_as_uint(f);
  u += 0x7fffu + ((u >> 16) & 1u);
  return (uint16_t)(u >> 16);
}

__device__ __forceinline__ u32x4 pack_bf16x8(floatx8 a) {
  u32x4 p;
  p.x = (uint32_t)f2bf(a[0]) | ((uint32_t)f2bf(a[1]) << 16);
  p.y = (uint32_t)f2bf(a[2]) | ((uint32_t)f2bf(a[3]) << 16);
  p.z = (uint32_t)f2bf(a[4]) | ((uint32_t)f2bf(a[5]) << 16);
  p.w = (uint32_t)f2bf(a[6]) | ((uint32_t)f2bf(a[7]) << 16);
  return p;
}

#define WMMA_BF16(A, B, C) \
  __builtin_amdgcn_wmma_f32_16x16x32_bf16(false, (A), false, (B), (short)0, (C), false, false)

// ---- elementwise fp32 -> bf16 ----------------------------------------------
__global__ void cvt_f32_bf16(const float* __restrict__ in,
                             uint16_t* __restrict__ out, int n) {
  int i = blockIdx.x * blockDim.x + threadIdx.x;
  if (i < n) out[i] = f2bf(in[i]);
}

// ---- block reduction helper -------------------------------------------------
__device__ __forceinline__ float block_reduce_sum(float v, float* red) {
  int t = threadIdx.x;
  red[t] = v; __syncthreads();
  for (int s = 128; s > 0; s >>= 1) { if (t < s) red[t] += red[t + s]; __syncthreads(); }
  float r = red[0];
  __syncthreads();
  return r;
}

// ---- spectral normalization of Wv (one workgroup, 256 threads) -------------
__global__ __launch_bounds__(256)
void spectral_norm_kernel(const float* __restrict__ W, const float* __restrict__ u0,
                          uint16_t* __restrict__ Wsn) {
  __shared__ float su[256], sv[256], red[256];
  int t = threadIdx.x;
  float ui = u0[t];
  float nu = sqrtf(block_reduce_sum(ui * ui, red)) + 1e-12f;
  su[t] = ui / nu; __syncthreads();
  for (int it = 0; it < 5; ++it) {
    float av = 0.f;
    for (int i = 0; i < 256; ++i) av += W[i * 256 + t] * su[i];
    float nv = sqrtf(block_reduce_sum(av * av, red)) + 1e-12f;
    sv[t] = av / nv; __syncthreads();
    float au = 0.f;
    for (int j = 0; j < 256; ++j) au += W[t * 256 + j] * sv[j];
    float nu2 = sqrtf(block_reduce_sum(au * au, red)) + 1e-12f;
    su[t] = au / nu2; __syncthreads();
  }
  float wv = 0.f;
  for (int j = 0; j < 256; ++j) wv += W[t * 256 + j] * sv[j];
  float sigma = block_reduce_sum(su[t] * wv, red);
  float inv = 1.0f / sigma;
  for (int j = t; j < 256 * 256; j += 256) Wsn[j] = f2bf(W[j] * inv);
}

// ---- q/k/v projections, 32x32 register-blocked tile per wave ---------------
// q,k stored row-major bf16 [B,N,H]; v stored TRANSPOSED bf16 [B,D,N]
__global__ __launch_bounds__(256)
void gemm_qkv(const uint16_t* __restrict__ xb,
              const uint16_t* __restrict__ Wqb, const uint16_t* __restrict__ Wkb,
              const uint16_t* __restrict__ Wvb,
              const float* __restrict__ bq, const float* __restrict__ bk,
              const float* __restrict__ bv,
              uint16_t* __restrict__ qo, uint16_t* __restrict__ ko,
              uint16_t* __restrict__ vT) {
  int w    = blockIdx.x * 8 + (threadIdx.x >> 5);
  int lane = threadIdx.x & 31;
  int mat  = w >> 13;                 // 8192 tiles per matrix
  int r    = w & 8191;
  int b    = r >> 6;
  int t    = r & 63;
  int mt   = t >> 3, nt = t & 7;      // 32x32 tiles: 8 x 8 grid
  const uint16_t* W    = (mat == 0) ? Wqb : (mat == 1) ? Wkb : Wvb;
  const float*    bias = (mat == 0) ? bq  : (mat == 1) ? bk  : bv;

  int li = lane & 15, half = lane >> 4;
  const uint16_t* xr0 = xb + ((size_t)b * Nseq + mt * 32 + li) * Dim;
  const uint16_t* xr1 = xr0 + (size_t)16 * Dim;
  const uint16_t* wr0 = W + (size_t)(nt * 32 + li) * Dim;
  const uint16_t* wr1 = wr0 + (size_t)16 * Dim;

  floatx8 acc[2][2] = {};
  for (int k0 = 0; k0 < Dim; k0 += 32) {
    Frag a0, a1, b0, b1;
    a0.q[0] = *(const u32x4*)(xr0 + k0 + half * 8);
    a0.q[1] = *(const u32x4*)(xr0 + k0 + half * 8 + 16);
    a1.q[0] = *(const u32x4*)(xr1 + k0 + half * 8);
    a1.q[1] = *(const u32x4*)(xr1 + k0 + half * 8 + 16);
    b0.q[0] = *(const u32x4*)(wr0 + k0 + half * 8);
    b0.q[1] = *(const u32x4*)(wr0 + k0 + half * 8 + 16);
    b1.q[0] = *(const u32x4*)(wr1 + k0 + half * 8);
    b1.q[1] = *(const u32x4*)(wr1 + k0 + half * 8 + 16);
    acc[0][0] = WMMA_BF16(a0.v, b0.v, acc[0][0]);
    acc[0][1] = WMMA_BF16(a0.v, b1.v, acc[0][1]);
    acc[1][0] = WMMA_BF16(a1.v, b0.v, acc[1][0]);
    acc[1][1] = WMMA_BF16(a1.v, b1.v, acc[1][1]);
  }

  if (mat == 2) {
    // transposed store: vT[b][d][m]; 8 acc rows per lane are contiguous in m
#pragma unroll
    for (int i = 0; i < 2; ++i)
#pragma unroll
      for (int j = 0; j < 2; ++j) {
        int col = nt * 32 + j * 16 + li;          // d
        float bia = bias[col];
        floatx8 v = acc[i][j];
#pragma unroll
        for (int rr = 0; rr < 8; ++rr) v[rr] += bia;
        size_t m0 = (size_t)mt * 32 + i * 16 + half * 8;
        *(u32x4*)(vT + ((size_t)b * Dim + col) * Nseq + m0) = pack_bf16x8(v);
      }
  } else {
    uint16_t* out = (mat == 0) ? qo : ko;
#pragma unroll
    for (int i = 0; i < 2; ++i)
#pragma unroll
      for (int j = 0; j < 2; ++j) {
        int col = nt * 32 + j * 16 + li;
        float bia = bias[col];
        uint16_t* orow = out + ((size_t)b * Nseq + mt * 32 + i * 16) * Hdim + col;
#pragma unroll
        for (int rr = 0; rr < 8; ++rr)
          orow[(size_t)(rr + half * 8) * Hdim] = f2bf(acc[i][j][rr] + bia);
      }
  }
}

// ---- scores = Q K^T per batch, 32x32 tile per wave, fp32 out ---------------
__global__ __launch_bounds__(256)
void gemm_scores(const uint16_t* __restrict__ qb, const uint16_t* __restrict__ kb,
                 float* __restrict__ scores) {
  int w    = blockIdx.x * 8 + (threadIdx.x >> 5);
  int lane = threadIdx.x & 31;
  int b  = w >> 6;
  int t  = w & 63;
  int mt = t >> 3, nt = t & 7;
  int li = lane & 15, half = lane >> 4;
  const uint16_t* qr0 = qb + ((size_t)b * Nseq + mt * 32 + li) * Hdim;
  const uint16_t* qr1 = qr0 + (size_t)16 * Hdim;
  const uint16_t* kr0 = kb + ((size_t)b * Nseq + nt * 32 + li) * Hdim;
  const uint16_t* kr1 = kr0 + (size_t)16 * Hdim;

  floatx8 acc[2][2] = {};
  for (int k0 = 0; k0 < Hdim; k0 += 32) {
    Frag a0, a1, b0, b1;
    a0.q[0] = *(const u32x4*)(qr0 + k0 + half * 8);
    a0.q[1] = *(const u32x4*)(qr0 + k0 + half * 8 + 16);
    a1.q[0] = *(const u32x4*)(qr1 + k0 + half * 8);
    a1.q[1] = *(const u32x4*)(qr1 + k0 + half * 8 + 16);
    b0.q[0] = *(const u32x4*)(kr0 + k0 + half * 8);
    b0.q[1] = *(const u32x4*)(kr0 + k0 + half * 8 + 16);
    b1.q[0] = *(const u32x4*)(kr1 + k0 + half * 8);
    b1.q[1] = *(const u32x4*)(kr1 + k0 + half * 8 + 16);
    acc[0][0] = WMMA_BF16(a0.v, b0.v, acc[0][0]);
    acc[0][1] = WMMA_BF16(a0.v, b1.v, acc[0][1]);
    acc[1][0] = WMMA_BF16(a1.v, b0.v, acc[1][0]);
    acc[1][1] = WMMA_BF16(a1.v, b1.v, acc[1][1]);
  }
#pragma unroll
  for (int i = 0; i < 2; ++i)
#pragma unroll
    for (int j = 0; j < 2; ++j) {
      float* orow = scores + ((size_t)b * Nseq + mt * 32 + i * 16) * Nseq
                           + nt * 32 + j * 16 + li;
#pragma unroll
      for (int rr = 0; rr < 8; ++rr)
        orow[(size_t)(rr + half * 8) * Nseq] = acc[i][j][rr];
    }
}

// ---- spectral radius via power iteration (one block per batch) -------------
__global__ __launch_bounds__(256)
void rho_power(const float* __restrict__ scores, float* __restrict__ inv_rho) {
  __shared__ float sx[256], red[256];
  int b = blockIdx.x, t = threadIdx.x;
  const float* S = scores + (size_t)b * Nseq * Nseq + (size_t)t * Nseq;
  sx[t] = 1.0f / 16.0f;
  __syncthreads();
  float rho = 1.0f;
  for (int it = 0; it < 32; ++it) {
    float y = 0.f;
    for (int m = 0; m < Nseq; ++m) y += S[m] * sx[m];
    rho = sqrtf(block_reduce_sum(y * y, red)) + 1e-30f;
    sx[t] = y / rho;
    __syncthreads();
  }
  if (t == 0) inv_rho[b] = 1.0f / rho;
}

// ---- attention = scores/rho (in place) + bf16 copy -------------------------
__global__ void scale_attn(float* __restrict__ attn, uint16_t* __restrict__ attn_bf,
                           const float* __restrict__ inv_rho) {
  size_t i = (size_t)blockIdx.x * blockDim.x + threadIdx.x;
  if (i < (size_t)Bsz * Nseq * Nseq) {
    int b = (int)(i >> 16);          // N*N == 65536
    float v = attn[i] * inv_rho[b];
    attn[i] = v;
    attn_bf[i] = f2bf(v);
  }
}

// ---- weighted = attention @ vals (vals pre-transposed: vT[B,D,N]) ----------
__global__ __launch_bounds__(256)
void gemm_weighted(const uint16_t* __restrict__ ab, const uint16_t* __restrict__ vT,
                   float* __restrict__ outw) {
  int w    = blockIdx.x * 8 + (threadIdx.x >> 5);
  int lane = threadIdx.x & 31;
  int b  = w >> 6;
  int t  = w & 63;
  int mt = t >> 3, nt = t & 7;
  int li = lane & 15, half = lane >> 4;
  const uint16_t* ar0 = ab + ((size_t)b * Nseq + mt * 32 + li) * Nseq;
  const uint16_t* ar1 = ar0 + (size_t)16 * Nseq;
  const uint16_t* vc0 = vT + ((size_t)b * Dim + nt * 32 + li) * Nseq; // column d
  const uint16_t* vc1 = vc0 + (size_t)16 * Nseq;

  floatx8 acc[2][2] = {};
  for (int k0 = 0; k0 < Nseq; k0 += 32) {
    Frag a0, a1, b0, b1;
    a0.q[0] = *(const u32x4*)(ar0 + k0 + half * 8);
    a0.q[1] = *(const u32x4*)(ar0 + k0 + half * 8 + 16);
    a1.q[0] = *(const u32x4*)(ar1 + k0 + half * 8);
    a1.q[1] = *(const u32x4*)(ar1 + k0 + half * 8 + 16);
    b0.q[0] = *(const u32x4*)(vc0 + k0 + half * 8);
    b0.q[1] = *(const u32x4*)(vc0 + k0 + half * 8 + 16);
    b1.q[0] = *(const u32x4*)(vc1 + k0 + half * 8);
    b1.q[1] = *(const u32x4*)(vc1 + k0 + half * 8 + 16);
    acc[0][0] = WMMA_BF16(a0.v, b0.v, acc[0][0]);
    acc[0][1] = WMMA_BF16(a0.v, b1.v, acc[0][1]);
    acc[1][0] = WMMA_BF16(a1.v, b0.v, acc[1][0]);
    acc[1][1] = WMMA_BF16(a1.v, b1.v, acc[1][1]);
  }
#pragma unroll
  for (int i = 0; i < 2; ++i)
#pragma unroll
    for (int j = 0; j < 2; ++j) {
      float* orow = outw + ((size_t)b * Nseq + mt * 32 + i * 16) * Dim
                         + nt * 32 + j * 16 + li;
#pragma unroll
      for (int rr = 0; rr < 8; ++rr)
        orow[(size_t)(rr + half * 8) * Dim] = acc[i][j][rr];
    }
}

// ---- launcher ---------------------------------------------------------------
extern "C" void kernel_launch(void* const* d_in, const int* in_sizes, int n_in,
                              void* d_out, int out_size, void* d_ws, size_t ws_size,
                              hipStream_t stream) {
  const float* x  = (const float*)d_in[0];
  const float* Wq = (const float*)d_in[1];
  const float* bq = (const float*)d_in[2];
  const float* Wk = (const float*)d_in[3];
  const float* bk = (const float*)d_in[4];
  const float* Wv = (const float*)d_in[5];
  const float* bv = (const float*)d_in[6];
  const float* u0 = (const float*)d_in[7];

  // workspace layout (~84 MB)
  char*  ws  = (char*)d_ws;
  size_t off = 0;
  auto alloc = [&](size_t bytes) -> void* {
    void* p = ws + off; off += (bytes + 255) & ~(size_t)255; return p;
  };
  const size_t XN = (size_t)Bsz * Nseq * Dim;          // 8,388,608
  uint16_t* xb   = (uint16_t*)alloc(XN * 2);
  uint16_t* Wqb  = (uint16_t*)alloc((size_t)Hdim * Dim * 2);
  uint16_t* Wkb  = (uint16_t*)alloc((size_t)Hdim * Dim * 2);
  uint16_t* Wvb  = (uint16_t*)alloc((size_t)Dim * Dim * 2);
  uint16_t* qo   = (uint16_t*)alloc(XN * 2);
  uint16_t* ko   = (uint16_t*)alloc(XN * 2);
  uint16_t* vT   = (uint16_t*)alloc(XN * 2);           // transposed vals [B,D,N]
  uint16_t* abf  = (uint16_t*)alloc((size_t)Bsz * Nseq * Nseq * 2);
  float*    invr = (float*)alloc(Bsz * sizeof(float));

  float* outW = (float*)d_out;                         // [B,N,D]
  float* outA = outW + XN;                             // [B,N,N]

  cvt_f32_bf16<<<(int)(XN / 256), 256, 0, stream>>>(x, xb, (int)XN);
  cvt_f32_bf16<<<256, 256, 0, stream>>>(Wq, Wqb, Hdim * Dim);
  cvt_f32_bf16<<<256, 256, 0, stream>>>(Wk, Wkb, Hdim * Dim);
  spectral_norm_kernel<<<1, 256, 0, stream>>>(Wv, u0, Wvb);

  gemm_qkv<<<3072, 256, 0, stream>>>(xb, Wqb, Wkb, Wvb, bq, bk, bv, qo, ko, vT);
  gemm_scores<<<1024, 256, 0, stream>>>(qo, ko, outA);
  rho_power<<<Bsz, 256, 0, stream>>>(outA, invr);
  scale_attn<<<32768, 256, 0, stream>>>(outA, abf, invr);
  gemm_weighted<<<1024, 256, 0, stream>>>(abf, vT, outW);
}
